// RelativeGlobalAttention_1984274890871
// MI455X (gfx1250) — compile-verified, matrix operand order checked by reading, not run on hardware
//
#include <hip/hip_runtime.h>
#include <hip/hip_bf16.h>
#include <stdint.h>

// ---------------------------------------------------------------------------
// Relative Global Attention for MI455X (gfx1250), wave32 + WMMA f16->f32,
// TDM (tensor_load_to_lds) staging of the Q tile in the fused attention kernel.
// Pipeline: cvt -> QKV proj (WMMA, 16x64/wave) ->
//           fused QE/QK^T+skew/softmax/AV (WMMA + TDM) -> output proj (WMMA).
// ---------------------------------------------------------------------------

#define B_ 4
#define S_ 1024
#define D_ 512
#define H_ 8
#define DH_ 64

typedef _Float16 half8  __attribute__((ext_vector_type(8)));
typedef _Float16 half16 __attribute__((ext_vector_type(16)));
typedef float    f32x8  __attribute__((ext_vector_type(8)));
typedef uint32_t u32x4  __attribute__((ext_vector_type(4)));
typedef int      i32x4  __attribute__((ext_vector_type(4)));
typedef int      i32x8  __attribute__((ext_vector_type(8)));

// Load a 16-half A/B fragment for v_wmma_f32_16x16x32_f16.
// lane L: sel = L/16; elems 0..7 -> K = kbase..kbase+7,
//         elems 8..15 -> K = kbase+16..kbase+23, kbase = kstep*32 + sel*8.
// p points at the lane's matrix row (A: row M; B: row of W^T / K / V^T).
__device__ __forceinline__ half16 frag_ld(const _Float16* __restrict__ p, int kbase) {
  half8 lo = *(const half8*)(p + kbase);
  half8 hi = *(const half8*)(p + kbase + 16);
  half16 r;
#pragma unroll
  for (int e = 0; e < 8; ++e) { r[e] = lo[e]; r[e + 8] = hi[e]; }
  return r;
}

__device__ __forceinline__ f32x8 wmma16(half16 a, half16 b, f32x8 c) {
  return __builtin_amdgcn_wmma_f32_16x16x32_f16(false, a, false, b, (short)0, c,
                                                false, false);
}

// TDM: stage a [rows x 64] f16 row-major tile (stride 64 elems) into LDS.
// D# built per CDNA5 ISA §8: group0 = {flags/count, lds_addr, gaddr lo/hi+type},
// group1 = {data_size, tensor dims/strides, tile dims}. 2D tile -> groups 2/3 zero.
__device__ __forceinline__ void tdm_load_tile16x64(const _Float16* gsrc,
                                                   _Float16* lds_dst) {
  uint64_t ga = (uint64_t)(uintptr_t)gsrc;
  uint32_t la = (uint32_t)(uintptr_t)lds_dst;   // LDS aperture: addr[31:0] = LDS byte offset
  u32x4 g0;
  g0[0] = 1u;                                   // count=1 (valid), all flags 0
  g0[1] = la;                                   // lds_addr
  g0[2] = (uint32_t)ga;                         // global_addr[31:0]
  g0[3] = (uint32_t)((ga >> 32) & 0x01FFFFFFu) | 0x80000000u;  // addr[56:32] | type=2
  i32x8 g1;
  g1[0] = 0x00010000;          // data_size = 1 -> 2 bytes
  g1[1] = (int)(64u << 16);    // tensor_dim0 = 64 (lo16 at bits[31:16])
  g1[2] = (int)(16u << 16);    // tensor_dim0 hi = 0 | tensor_dim1 lo16 = 16
  g1[3] = (int)(64u << 16);    // tensor_dim1 hi = 0 | tile_dim0 = 64
  g1[4] = 16;                  // tile_dim1 = 16, tile_dim2 = 0 (2D)
  g1[5] = 64;                  // tensor_dim0_stride[31:0] = 64
  g1[6] = 0;                   // stride0 hi | stride1 lo
  g1[7] = 0;
  i32x4 gz = {0, 0, 0, 0};
#if __clang_major__ >= 23
  i32x8 gz8 = {0, 0, 0, 0, 0, 0, 0, 0};
  __builtin_amdgcn_tensor_load_to_lds(g0, g1, gz, gz, gz8, 0);
#else
  __builtin_amdgcn_tensor_load_to_lds(g0, g1, gz, gz, 0);
#endif
  __builtin_amdgcn_s_wait_tensorcnt(0);
}

// ---------------------------------------------------------------------------
// Elementwise f32 -> f16 convert
__global__ void k_cvt(const float* __restrict__ src, _Float16* __restrict__ dst,
                      int n) {
  int i = blockIdx.x * blockDim.x + threadIdx.x;
  if (i < n) dst[i] = (_Float16)src[i];
}

// Transposing convert: dst[n*K + k] = src[k*N + n]  (store W^T for B-fragments)
__global__ void k_cvtT(const float* __restrict__ src, _Float16* __restrict__ dst,
                       int K, int N) {
  int i = blockIdx.x * blockDim.x + threadIdx.x;
  if (i < K * N) {
    int k = i / N, n = i % N;
    dst[(size_t)n * K + k] = (_Float16)src[i];
  }
}

// ---------------------------------------------------------------------------
// QKV projection: Y[s,n] = sum_k X[s,k] * W[k,n] + bias[n]
// One wave per 16x64 output tile (4 accumulators share each A fragment).
// mode 0: dst = head-split [B][H][S][DH]   (Q, K)
// mode 1: dst = head-split transposed [B][H][DH][S]  (V^T)
__global__ __launch_bounds__(32) void k_proj(const _Float16* __restrict__ X,
                                             const _Float16* __restrict__ WT,
                                             const float* __restrict__ bias,
                                             _Float16* __restrict__ dst,
                                             int mode) {
  const int i0 = blockIdx.x * 16;
  const int n0 = blockIdx.y * 64;
  const int lane = threadIdx.x & 31;
  const int sel = lane >> 4, ln = lane & 15;
  const _Float16* pA = X + (size_t)(i0 + ln) * D_;
  const _Float16* pB[4];
#pragma unroll
  for (int c = 0; c < 4; ++c) pB[c] = WT + (size_t)(n0 + c * 16 + ln) * D_;
  f32x8 acc[4] = {};
#pragma unroll
  for (int ks = 0; ks < D_ / 32; ++ks) {
    int kb = ks * 32 + sel * 8;
    half16 a = frag_ld(pA, kb);
#pragma unroll
    for (int c = 0; c < 4; ++c) {
      __builtin_prefetch(pB[c] + kb + 64, 0, 1);
      acc[c] = wmma16(a, frag_ld(pB[c], kb), acc[c]);
    }
  }
#pragma unroll
  for (int c = 0; c < 4; ++c) {
    const int col = n0 + c * 16 + ln;
    const float bval = bias[col];
    const int hh = col >> 6, dh = col & 63;
#pragma unroll
    for (int v = 0; v < 8; ++v) {
      int row = i0 + v + 8 * sel;          // global M row = b*S + s
      int bidx = row >> 10, s = row & 1023;
      float val = acc[c][v] + bval;
      if (mode == 0)
        dst[(size_t)(((bidx * H_ + hh) * S_) + s) * DH_ + dh] = (_Float16)val;
      else
        dst[(size_t)(((bidx * H_ + hh) * DH_) + dh) * S_ + s] = (_Float16)val;
    }
  }
}

// ---------------------------------------------------------------------------
// Output projection: out[s,n] = sum_k ctx[s,k]*Wo[k,n] + bo[n]  (f32 out)
__global__ __launch_bounds__(32) void k_oproj(const _Float16* __restrict__ X,
                                              const _Float16* __restrict__ WT,
                                              const float* __restrict__ bias,
                                              float* __restrict__ out) {
  const int i0 = blockIdx.x * 16;
  const int n0 = blockIdx.y * 64;
  const int lane = threadIdx.x & 31;
  const int sel = lane >> 4, ln = lane & 15;
  const _Float16* pA = X + (size_t)(i0 + ln) * D_;
  const _Float16* pB[4];
#pragma unroll
  for (int c = 0; c < 4; ++c) pB[c] = WT + (size_t)(n0 + c * 16 + ln) * D_;
  f32x8 acc[4] = {};
#pragma unroll
  for (int ks = 0; ks < D_ / 32; ++ks) {
    int kb = ks * 32 + sel * 8;
    half16 a = frag_ld(pA, kb);
#pragma unroll
    for (int c = 0; c < 4; ++c)
      acc[c] = wmma16(a, frag_ld(pB[c], kb), acc[c]);
  }
#pragma unroll
  for (int c = 0; c < 4; ++c) {
    const int col = n0 + c * 16 + ln;
    const float bval = bias[col];
#pragma unroll
    for (int v = 0; v < 8; ++v) {
      int row = i0 + v + 8 * sel;
      out[(size_t)row * D_ + col] = acc[c][v] + bval;
    }
  }
}

// ---------------------------------------------------------------------------
// Fused attention: one workgroup (8 waves, 256 thr) per (b, h, 16 query rows).
//  stage 0: TDM-stage the 16x64 Q tile into LDS (tensor_load_to_lds).
//  stage 1: QE[i, m] = q_i . e_m  via WMMA -> LDS f16 (16 x 1024)
//  stage 2: logits[i,j] = (q_i.k_j + Srel)/8 via WMMA, Srel = QE[i,1023+j-i]
//           for j<=i else 0 (== reference mask+skew). Written to attn region.
//  stage 3: in-place row softmax over attn (final attn output), f16 probs
//           cached back into LDS.
//  stage 4: ctx = attn @ V via WMMA (A from LDS, B from V^T), f16 ctx out.
__global__ __launch_bounds__(256) void k_att(const _Float16* __restrict__ Q,
                                             const _Float16* __restrict__ K,
                                             const _Float16* __restrict__ VT,
                                             const _Float16* __restrict__ E16,
                                             float* __restrict__ attn,
                                             _Float16* __restrict__ ctx) {
  __shared__ _Float16 qtile[16 * DH_];  // 2 KB: TDM-staged Q tile
  __shared__ _Float16 attq[16 * S_];    // 32 KB: QE, then attn probs (f16)
  __shared__ float    red[256];         // softmax row reductions

  const int bt = blockIdx.x;
  const int it = bt & 63;
  const int h  = (bt >> 6) & 7;
  const int b  = bt >> 9;
  const int i0 = it * 16;

  const int tid  = threadIdx.x;
  const int wid  = tid >> 5;
  const int lane = tid & 31;
  const int sel  = lane >> 4, ln = lane & 15;

  const _Float16* qsrc = Q + ((size_t)((b * H_ + h) * S_) + i0) * DH_;
  const _Float16* kh   = K + (size_t)((b * H_ + h) * S_) * DH_;
  float* lgbase = attn + (size_t)((b * H_ + h) * S_ + i0) * S_;

  // ---- stage 0: TDM Q-tile -> LDS (one wave issues; TENSORcnt; barrier)
  if (wid == 0) tdm_load_tile16x64(qsrc, qtile);
  __syncthreads();
  const _Float16* pA = qtile + ln * DH_;   // A fragment rows now in LDS

  // ---- stage 1: QE tiles (waves stride over 64 m-tiles)
  for (int jt = wid; jt < 64; jt += 8) {
    const int m0 = jt * 16;
    const _Float16* pB = E16 + (size_t)(m0 + ln) * DH_;
    f32x8 acc = {};
#pragma unroll
    for (int ks = 0; ks < 2; ++ks) {
      int kb = ks * 32 + sel * 8;
      acc = wmma16(frag_ld(pA, kb), frag_ld(pB, kb), acc);
    }
#pragma unroll
    for (int v = 0; v < 8; ++v) {
      int il = v + 8 * sel;
      attq[il * S_ + m0 + ln] = (_Float16)acc[v];
    }
  }
  __syncthreads();

  // ---- stage 2: logits = (QK^T + skewed QE) * 1/sqrt(DH)
  for (int jt = wid; jt < 64; jt += 8) {
    const int j0 = jt * 16;
    const _Float16* pB = kh + (size_t)(j0 + ln) * DH_;
    __builtin_prefetch(pB + 16 * DH_, 0, 1);
    f32x8 acc = {};
#pragma unroll
    for (int ks = 0; ks < 2; ++ks) {
      int kb = ks * 32 + sel * 8;
      acc = wmma16(frag_ld(pA, kb), frag_ld(pB, kb), acc);
    }
#pragma unroll
    for (int v = 0; v < 8; ++v) {
      int il = v + 8 * sel;
      int i = i0 + il;
      int j = j0 + ln;
      float sr = (j <= i) ? (float)attq[il * S_ + (S_ - 1 + j - i)] : 0.0f;
      lgbase[(size_t)il * S_ + j] = (acc[v] + sr) * 0.125f;
    }
  }
  __threadfence_block();   // make stage-2 global stores visible across waves
  __syncthreads();

  // ---- stage 3: softmax (16 rows x 16 threads/row), in place in attn
  {
    const int rid = tid >> 4, lr = tid & 15;
    float* row = lgbase + (size_t)rid * S_;
    float mx = -3.4e38f;
    for (int j = lr; j < S_; j += 16) mx = fmaxf(mx, row[j]);
    red[rid * 16 + lr] = mx;
    __syncthreads();
    float rowmax = -3.4e38f;
#pragma unroll
    for (int k2 = 0; k2 < 16; ++k2) rowmax = fmaxf(rowmax, red[rid * 16 + k2]);
    __syncthreads();
    float s = 0.0f;
    for (int j = lr; j < S_; j += 16) {
      float p = __expf(row[j] - rowmax);
      row[j] = p;
      s += p;
    }
    red[rid * 16 + lr] = s;
    __syncthreads();
    float rs = 0.0f;
#pragma unroll
    for (int k2 = 0; k2 < 16; ++k2) rs += red[rid * 16 + k2];
    const float inv = 1.0f / rs;
    for (int j = lr; j < S_; j += 16) {
      float val = row[j] * inv;
      row[j] = val;                        // final attn output (f32)
      attq[rid * S_ + j] = (_Float16)val;  // f16 copy for AV matmul
    }
  }
  __syncthreads();

  // ---- stage 4: ctx = attn @ V  (waves 0..3 each own a 16-wide DH slice)
  if (wid < 4) {
    const int n0 = wid * 16;
    const _Float16* pBv = VT + (size_t)((b * H_ + h) * DH_ + n0 + ln) * S_;
    f32x8 acc = {};
    for (int ms = 0; ms < S_ / 32; ++ms) {
      int kb = ms * 32 + sel * 8;
      half16 a;
#pragma unroll
      for (int e = 0; e < 8; ++e) {        // A fragment from LDS probs
        a[e]     = attq[ln * S_ + kb + e];
        a[e + 8] = attq[ln * S_ + kb + 16 + e];
      }
      acc = wmma16(a, frag_ld(pBv, kb), acc);
    }
#pragma unroll
    for (int v = 0; v < 8; ++v) {
      int il = v + 8 * sel;
      int sg = i0 + il;
      int col = h * DH_ + n0 + ln;
      ctx[(size_t)(b * S_ + sg) * D_ + col] = (_Float16)acc[v];
    }
  }
}

// ---------------------------------------------------------------------------
extern "C" void kernel_launch(void* const* d_in, const int* in_sizes, int n_in,
                              void* d_out, int out_size, void* d_ws,
                              size_t ws_size, hipStream_t stream) {
  (void)in_sizes; (void)n_in; (void)out_size; (void)ws_size;

  const float* q_in = (const float*)d_in[0];
  const float* k_in = (const float*)d_in[1];
  const float* v_in = (const float*)d_in[2];
  const float* Wq = (const float*)d_in[3];
  const float* bq = (const float*)d_in[4];
  const float* Wk = (const float*)d_in[5];
  const float* bk = (const float*)d_in[6];
  const float* Wv = (const float*)d_in[7];
  const float* bv = (const float*)d_in[8];
  const float* E  = (const float*)d_in[9];
  const float* Wo = (const float*)d_in[10];
  const float* bo = (const float*)d_in[11];

  constexpr size_t NX = (size_t)B_ * S_ * D_;   // 2,097,152 (also B*H*S*DH)
  constexpr size_t NW = (size_t)D_ * D_;        // 262,144
  constexpr size_t NE = (size_t)S_ * DH_;       // 65,536

  _Float16* ws = (_Float16*)d_ws;
  _Float16* XQ   = ws;
  _Float16* XK   = XQ + NX;
  _Float16* XV   = XK + NX;
  _Float16* Q16  = XV + NX;
  _Float16* K16  = Q16 + NX;
  _Float16* VT16 = K16 + NX;
  _Float16* CTX  = VT16 + NX;
  _Float16* WQT  = CTX + NX;
  _Float16* WKT  = WQT + NW;
  _Float16* WVT  = WKT + NW;
  _Float16* WOT  = WVT + NW;
  _Float16* E16  = WOT + NW;

  float* out  = (float*)d_out;
  float* attn = out + NX;   // (B,H,S,S) follows (B,S,D)

  // Converts
  k_cvt<<<NX / 256, 256, 0, stream>>>(q_in, XQ, (int)NX);
  k_cvt<<<NX / 256, 256, 0, stream>>>(k_in, XK, (int)NX);
  k_cvt<<<NX / 256, 256, 0, stream>>>(v_in, XV, (int)NX);
  k_cvtT<<<NW / 256, 256, 0, stream>>>(Wq, WQT, D_, D_);
  k_cvtT<<<NW / 256, 256, 0, stream>>>(Wk, WKT, D_, D_);
  k_cvtT<<<NW / 256, 256, 0, stream>>>(Wv, WVT, D_, D_);
  k_cvtT<<<NW / 256, 256, 0, stream>>>(Wo, WOT, D_, D_);
  // e = E[MAX_SEQ - S :, :]  -> rows 1024..2047
  k_cvt<<<NE / 256, 256, 0, stream>>>(E + NE, E16, (int)NE);

  // Projections (Q/K head-split, V transposed per head)
  dim3 pg(B_ * S_ / 16, D_ / 64);
  k_proj<<<pg, 32, 0, stream>>>(XQ, WQT, bq, Q16, 0);
  k_proj<<<pg, 32, 0, stream>>>(XK, WKT, bk, K16, 0);
  k_proj<<<pg, 32, 0, stream>>>(XV, WVT, bv, VT16, 1);

  // Fused attention (TDM Q-stage + QE + skew + QK^T + softmax + AV)
  k_att<<<B_ * H_ * (S_ / 16), 256, 0, stream>>>(Q16, K16, VT16, E16, attn, CTX);

  // Output projection
  k_oproj<<<pg, 32, 0, stream>>>(CTX, WOT, bo, out);
}
